// INDIG_43026982371946
// MI455X (gfx1250) — matrix operand order repaired and checked by reading, not compiled, not run on hardware
//
#include <hip/hip_runtime.h>
#include <math.h>

typedef _Float16 half_t;
typedef half_t v16h __attribute__((ext_vector_type(16)));
typedef half_t v8h  __attribute__((ext_vector_type(8)));
typedef float  v8f  __attribute__((ext_vector_type(8)));

#define D 128
#define N_GROUPS 4096
#define M_MEM 16
#define I_ITEMS 32

// ---------------- WMMA fragment loaders (16x16x32 f16) ----------------
// A matrix 16x32 (f16): lanes 0-15 row M=lane, K=0..15 contiguous;
// lanes 16-31 row M=lane-16, K=16..31 contiguous (matches B).
__device__ inline v16h load_a_frag(const half_t* src, int ld, int k0, int lane) {
  int row = lane & 15;
  int kb  = k0 + ((lane & 16) ? 16 : 0);
  const half_t* p = src + row * ld + kb;
  return *(const v16h*)p;
}

// B matrix 32x16 (f16): lane holds one column (N = lane&15); lanes 0-15 hold
// K=0..15 contiguous, lanes 16-31 hold K=16..31 (this code's order, same as A).
// Column n of B == row n of the row-major weight W[NC][KC] (since out = A @ W^T).
__device__ inline v16h load_b_frag(const half_t* w, int ld, int n0, int k0, int lane) {
  const half_t* p = w + (size_t)(n0 + (lane & 15)) * ld + k0 + ((lane & 16) ? 16 : 0);
  return *(const v16h*)p;
}

// ---------------- Preprocessing: fixed query -> qk vector ----------------
__global__ __launch_bounds__(128) void INDIG_prep_kernel(
    const float* __restrict__ funw, const float* __restrict__ wq,
    const float* __restrict__ bq, const float* __restrict__ wk,
    const float* __restrict__ bk, float* __restrict__ qk_s, float* __restrict__ qb_s) {
  __shared__ float fw[D], q[D], red[D];
  int t = threadIdx.x;
  float v = funw[t];
  red[t] = v * v;
  __syncthreads();
  for (int s = 64; s > 0; s >>= 1) { if (t < s) red[t] += red[t + s]; __syncthreads(); }
  float nrm = fmaxf(sqrtf(red[0]), 1e-12f);
  fw[t] = v / nrm;
  __syncthreads();
  float acc = bq[t];
  for (int d = 0; d < D; ++d) acc += fw[d] * wq[t * D + d];
  q[t] = acc;
  __syncthreads();
  const float inv_sqrtD = 0.0883883476483184f; // 1/sqrt(128)
  float a2 = 0.f;
  for (int j = 0; j < D; ++j) a2 += q[j] * wk[j * D + t];
  qk_s[t] = a2 * inv_sqrtD;
  if (t == 0) {
    float s = 0.f;
    for (int j = 0; j < D; ++j) s += q[j] * bk[j];
    *qb_s = s * inv_sqrtD;
  }
}

// ---------------- f32 -> f16 weight conversion ----------------
__global__ void INDIG_cvt_kernel(const float* __restrict__ src,
                                 half_t* __restrict__ dst, int n) {
  int i = blockIdx.x * blockDim.x + threadIdx.x;
  if (i < n) dst[i] = (half_t)src[i];
}

// ---------------- Phase 1: gather + mean + LayerNorm + attention pooling ----
// One block per group; 8 waves, 2 members each. Dominant cost: L2-resident
// gathers of item_emb_mf2 rows (coalesced float4 per lane, uniform indices
// become scalar loads).
__global__ __launch_bounds__(256) void INDIG_group_pool_kernel(
    const float* __restrict__ user_emb, const float* __restrict__ item_mf2,
    const float* __restrict__ ln_g, const float* __restrict__ ln_b,
    const int* __restrict__ group_members, const int* __restrict__ user_items,
    const float* __restrict__ qk_s, const float* __restrict__ qb_s,
    half_t* __restrict__ P) {
  __shared__ float mem2[M_MEM * D];
  __shared__ float scores[M_MEM];
  __shared__ float attn[M_MEM];
  int g = blockIdx.x;
  int tid = threadIdx.x, lane = tid & 31, wave = tid >> 5;
  float qb = *qb_s;
  float4 qk4 = *(const float4*)(qk_s + lane * 4);
  float4 lg = *(const float4*)(ln_g + lane * 4);
  float4 lb = *(const float4*)(ln_b + lane * 4);

  for (int mm = 0; mm < 2; ++mm) {
    int m = wave * 2 + mm;
    int uid = group_members[g * M_MEM + m];
    const int* uit = user_items + (size_t)uid * I_ITEMS;
    float4 acc = {0.f, 0.f, 0.f, 0.f};
#pragma unroll 4
    for (int it = 0; it < I_ITEMS; ++it) {
      int iid = uit[it];
      float4 v = *(const float4*)(item_mf2 + (size_t)iid * D + lane * 4);
      acc.x += v.x; acc.y += v.y; acc.z += v.z; acc.w += v.w;
    }
    float4 ue = *(const float4*)(user_emb + (size_t)uid * D + lane * 4);
    const float invI = 1.f / (float)I_ITEMS;
    acc.x = acc.x * invI + ue.x; acc.y = acc.y * invI + ue.y;
    acc.z = acc.z * invI + ue.z; acc.w = acc.w * invI + ue.w;
    // LayerNorm over D=128 (32 lanes x 4)
    float s  = acc.x + acc.y + acc.z + acc.w;
    float ss = acc.x * acc.x + acc.y * acc.y + acc.z * acc.z + acc.w * acc.w;
#pragma unroll
    for (int mk = 16; mk; mk >>= 1) {
      s  += __shfl_xor(s, mk, 32);
      ss += __shfl_xor(ss, mk, 32);
    }
    float mean = s * (1.f / D);
    float var  = ss * (1.f / D) - mean * mean;
    float rstd = rsqrtf(var + 1e-6f);
    float y0 = (acc.x - mean) * rstd * lg.x + lb.x;
    float y1 = (acc.y - mean) * rstd * lg.y + lb.y;
    float y2 = (acc.z - mean) * rstd * lg.z + lb.z;
    float y3 = (acc.w - mean) * rstd * lg.w + lb.w;
    int d0 = m * D + lane * 4;
    mem2[d0 + 0] = y0; mem2[d0 + 1] = y1; mem2[d0 + 2] = y2; mem2[d0 + 3] = y3;
    // score = (wk^T q)·mem2 / sqrt(D) + q·bk/sqrt(D)   (K never materialized)
    float sp = y0 * qk4.x + y1 * qk4.y + y2 * qk4.z + y3 * qk4.w;
#pragma unroll
    for (int mk = 16; mk; mk >>= 1) sp += __shfl_xor(sp, mk, 32);
    if (lane == 0) scores[m] = sp + qb;
  }
  __syncthreads();
  if (tid < M_MEM) {
    float mx = -1e30f;
    for (int i = 0; i < M_MEM; ++i) mx = fmaxf(mx, scores[i]);
    float sum = 0.f;
    for (int i = 0; i < M_MEM; ++i) sum += __expf(scores[i] - mx);
    attn[tid] = __expf(scores[tid] - mx) / sum;
  }
  __syncthreads();
  // pooled p (softmax weights sum to 1, so V collapses: ao = p @ wv^T + bv)
  if (tid < D) {
    float p = 0.f;
#pragma unroll
    for (int m = 0; m < M_MEM; ++m) p += attn[m] * mem2[m * D + tid];
    P[(size_t)g * D + tid] = (half_t)p;
  }
}

// ---------------- Phase 2: batched dense chain via WMMA (16 groups/block) ---
__global__ __launch_bounds__(128) void INDIG_dense_head_kernel(
    const half_t* __restrict__ P,
    const half_t* __restrict__ wv_h, const float* __restrict__ bv,
    const half_t* __restrict__ wo_h, const float* __restrict__ bo,
    const half_t* __restrict__ mw1_h, const float* __restrict__ mb1,
    const half_t* __restrict__ mw2_h, const float* __restrict__ mb2,
    const float* __restrict__ pw1, const float* __restrict__ pb1,
    const float* __restrict__ pw2, const float* __restrict__ pb2,
    const float* __restrict__ item_emb, const int* __restrict__ item_inputs,
    float* __restrict__ out) {
  __shared__ __align__(16) half_t bufA[16 * 128];
  __shared__ __align__(16) half_t bufT[16 * 128];
  __shared__ __align__(16) half_t bufH[16 * 256];
  __shared__ __align__(16) float  zbuf[16 * 128];
  int g0 = blockIdx.x * 16;
  int tid = threadIdx.x, lane = tid & 31, wave = tid >> 5;

  // load 16x128 f16 A tile (pooled p rows for 16 groups)
  {
    const uint4* src = (const uint4*)(P + (size_t)g0 * D);
    uint4* dst = (uint4*)bufA;
    for (int i = tid; i < 16 * 128 / 8; i += 128) dst[i] = src[i];
  }
  __syncthreads();

  // GEMM1: T = A @ wv^T + bv  -> bufT (f16)
  for (int nt = wave; nt < 8; nt += 4) {
    v8f acc = {};
#pragma unroll
    for (int kt = 0; kt < 4; ++kt) {
      v16h a = load_a_frag(bufA, 128, kt * 32, lane);
      v16h b = load_b_frag(wv_h, 128, nt * 16, kt * 32, lane);
      acc = __builtin_amdgcn_wmma_f32_16x16x32_f16(false, a, false, b, (short)0, acc, false, false);
    }
    int col = nt * 16 + (lane & 15);
    int rb = (lane & 16) ? 8 : 0;
    float bb = bv[col];
#pragma unroll
    for (int r = 0; r < 8; ++r) bufT[(rb + r) * 128 + col] = (half_t)(acc[r] + bb);
  }
  __syncthreads();

  // GEMM2: A2 = T @ wo^T + bo -> bufA (f16)
  for (int nt = wave; nt < 8; nt += 4) {
    v8f acc = {};
#pragma unroll
    for (int kt = 0; kt < 4; ++kt) {
      v16h a = load_a_frag(bufT, 128, kt * 32, lane);
      v16h b = load_b_frag(wo_h, 128, nt * 16, kt * 32, lane);
      acc = __builtin_amdgcn_wmma_f32_16x16x32_f16(false, a, false, b, (short)0, acc, false, false);
    }
    int col = nt * 16 + (lane & 15);
    int rb = (lane & 16) ? 8 : 0;
    float bb = bo[col];
#pragma unroll
    for (int r = 0; r < 8; ++r) bufA[(rb + r) * 128 + col] = (half_t)(acc[r] + bb);
  }
  __syncthreads();

  // GEMM3: H = relu(A2 @ mw1^T + mb1) -> bufH (16x256 f16)
  for (int nt = wave; nt < 16; nt += 4) {
    v8f acc = {};
#pragma unroll
    for (int kt = 0; kt < 4; ++kt) {
      v16h a = load_a_frag(bufA, 128, kt * 32, lane);
      v16h b = load_b_frag(mw1_h, 128, nt * 16, kt * 32, lane);
      acc = __builtin_amdgcn_wmma_f32_16x16x32_f16(false, a, false, b, (short)0, acc, false, false);
    }
    int col = nt * 16 + (lane & 15);
    int rb = (lane & 16) ? 8 : 0;
    float bb = mb1[col];
#pragma unroll
    for (int r = 0; r < 8; ++r)
      bufH[(rb + r) * 256 + col] = (half_t)fmaxf(acc[r] + bb, 0.f);
  }
  __syncthreads();

  // GEMM4: Z = H @ mw2^T + mb2 -> zbuf (f32)
  for (int nt = wave; nt < 8; nt += 4) {
    v8f acc = {};
#pragma unroll
    for (int kt = 0; kt < 8; ++kt) {
      v16h a = load_a_frag(bufH, 256, kt * 32, lane);
      v16h b = load_b_frag(mw2_h, 256, nt * 16, kt * 32, lane);
      acc = __builtin_amdgcn_wmma_f32_16x16x32_f16(false, a, false, b, (short)0, acc, false, false);
    }
    int col = nt * 16 + (lane & 15);
    int rb = (lane & 16) ? 8 : 0;
    float bb = mb2[col];
#pragma unroll
    for (int r = 0; r < 8; ++r) zbuf[(rb + r) * 128 + col] = acc[r] + bb;
  }
  __syncthreads();

  // Tail: L2-norm + NCF head, one wave per 4 rows (VALU, tiny)
  for (int rr = 0; rr < 4; ++rr) {
    int row = wave * 4 + rr;
    int g = g0 + row;
    int iid = item_inputs[g];
    float4 ie = *(const float4*)(item_emb + (size_t)iid * D + lane * 4);
    float4 zv = *(const float4*)(zbuf + row * 128 + lane * 4);
    float ss = zv.x * zv.x + zv.y * zv.y + zv.z * zv.z + zv.w * zv.w;
#pragma unroll
    for (int mk = 16; mk; mk >>= 1) ss += __shfl_xor(ss, mk, 32);
    float rn = 1.f / fmaxf(sqrtf(ss), 1e-12f);
    float zn[4]  = {zv.x * rn, zv.y * rn, zv.z * rn, zv.w * rn};
    float iev[4] = {ie.x, ie.y, ie.z, ie.w};
    float acc8[8];
#pragma unroll
    for (int o = 0; o < 8; ++o) acc8[o] = 0.f;
#pragma unroll
    for (int i = 0; i < 4; ++i) {
      int d = lane * 4 + i;
      float zi = zn[i] * iev[i];
#pragma unroll
      for (int o = 0; o < 8; ++o)
        acc8[o] += zi * pw1[o * 384 + d] + zn[i] * pw1[o * 384 + 128 + d]
                 + iev[i] * pw1[o * 384 + 256 + d];
    }
#pragma unroll
    for (int o = 0; o < 8; ++o) {
      float r = acc8[o];
#pragma unroll
      for (int mk = 16; mk; mk >>= 1) r += __shfl_xor(r, mk, 32);
      acc8[o] = r;
    }
    if (lane == 0) {
      float o0 = pb2[0], o1 = pb2[1];
#pragma unroll
      for (int o = 0; o < 8; ++o) {
        float h = fmaxf(acc8[o] + pb1[o], 0.f);
        o0 += h * pw2[o];
        o1 += h * pw2[8 + o];
      }
      out[g] = 1.f / (1.f + expf(-o0));                     // y_mu
      float sp = (o1 > 20.f) ? o1 : log1pf(expf(o1));       // softplus
      out[N_GROUPS + g] = 0.1f + 0.9f * sp;                 // y_sigma
    }
  }
}

extern "C" void kernel_launch(void* const* d_in, const int* in_sizes, int n_in,
                              void* d_out, int out_size, void* d_ws, size_t ws_size,
                              hipStream_t stream) {
  const float* funw      = (const float*)d_in[0];
  const float* user_emb  = (const float*)d_in[1];
  const float* item_emb  = (const float*)d_in[2];
  const float* item_mf2  = (const float*)d_in[3];
  const float* ln_g      = (const float*)d_in[4];
  const float* ln_b      = (const float*)d_in[5];
  const float* wq        = (const float*)d_in[6];
  const float* bq        = (const float*)d_in[7];
  const float* wk        = (const float*)d_in[8];
  const float* bk        = (const float*)d_in[9];
  const float* wv        = (const float*)d_in[10];
  const float* bv        = (const float*)d_in[11];
  const float* wo        = (const float*)d_in[12];
  const float* bo        = (const float*)d_in[13];
  const float* mw1       = (const float*)d_in[14];
  const float* mb1       = (const float*)d_in[15];
  const float* mw2       = (const float*)d_in[16];
  const float* mb2       = (const float*)d_in[17];
  const float* pw1       = (const float*)d_in[18];
  const float* pb1       = (const float*)d_in[19];
  const float* pw2       = (const float*)d_in[20];
  const float* pb2       = (const float*)d_in[21];
  const int* group_members = (const int*)d_in[22];
  const int* user_items    = (const int*)d_in[23];
  const int* item_inputs   = (const int*)d_in[24];

  char* ws = (char*)d_ws;
  float*  qk_s  = (float*)(ws);                 // 128 f32
  float*  qb_s  = (float*)(ws + 512);           // 1 f32
  half_t* wv_h  = (half_t*)(ws + 1024);         // 16384 f16
  half_t* wo_h  = wv_h + 16384;                 // 16384 f16
  half_t* mw1_h = wo_h + 16384;                 // 32768 f16
  half_t* mw2_h = mw1_h + 32768;                // 32768 f16
  half_t* Pbuf  = mw2_h + 32768;                // 4096*128 f16

  INDIG_prep_kernel<<<1, 128, 0, stream>>>(funw, wq, bq, wk, bk, qk_s, qb_s);
  INDIG_cvt_kernel<<<64, 256, 0, stream>>>(wv, wv_h, 16384);
  INDIG_cvt_kernel<<<64, 256, 0, stream>>>(wo, wo_h, 16384);
  INDIG_cvt_kernel<<<128, 256, 0, stream>>>(mw1, mw1_h, 32768);
  INDIG_cvt_kernel<<<128, 256, 0, stream>>>(mw2, mw2_h, 32768);
  INDIG_group_pool_kernel<<<N_GROUPS, 256, 0, stream>>>(
      user_emb, item_mf2, ln_g, ln_b, group_members, user_items, qk_s, qb_s, Pbuf);
  INDIG_dense_head_kernel<<<N_GROUPS / 16, 128, 0, stream>>>(
      Pbuf, wv_h, bv, wo_h, bo, mw1_h, mb1, mw2_h, mb2,
      pw1, pb1, pw2, pb2, item_emb, item_inputs, (float*)d_out);
}
